// LSTM_Orth_5909875000082
// MI455X (gfx1250) — matrix operand order repaired and checked by reading, not compiled, not run on hardware
//
#include <hip/hip_runtime.h>

// ---------------------------------------------------------------------------
// Fused LSTM cell for MI455X (gfx1250, wave32, WMMA + async global->LDS).
// Pass 1: convert x, h, and the 8 weight matrices f32 -> bf16 into d_ws.
// Pass 2: gates = x@Wx^T + h@Wh^T + b via v_wmma_f32_16x16x32_bf16 with
//         double-buffered GLOBAL_LOAD_ASYNC_TO_LDS_B128 staging, fused
//         sigmoid/tanh epilogue writing h_next / c_next.
// ---------------------------------------------------------------------------

typedef __attribute__((ext_vector_type(16))) __bf16 v16bf;
typedef __attribute__((ext_vector_type(8)))  float  v8f;
typedef __attribute__((ext_vector_type(4)))  int    v4i;
typedef unsigned short u16;

#define TILE_M 128
#define TILE_N 64          // gate-columns per block (per gate)
#define TILE_K 32
#define LDS_STRIDE 40      // 32 + 8 pad (u16 units) -> conflict-free frags
#define A_BUF (TILE_M * LDS_STRIDE)        // 5120 u16 per buffer
#define B_BUF (4 * TILE_N * LDS_STRIDE)    // 10240 u16 per buffer

#if __has_builtin(__builtin_amdgcn_global_load_async_to_lds_b128)
#define HAVE_ASYNC 1
#else
#define HAVE_ASYNC 0
#endif

__device__ __forceinline__ unsigned f2bf(float f) {
    unsigned u = __float_as_uint(f);
    return (u + 0x7FFFu + ((u >> 16) & 1u)) >> 16;   // round-to-nearest-even
}
__device__ __forceinline__ unsigned pack2(float lo, float hi) {
    return f2bf(lo) | (f2bf(hi) << 16);
}
__device__ __forceinline__ float fast_sigmoid(float v) {
    return 1.0f / (1.0f + __expf(-v));
}
__device__ __forceinline__ float fast_tanh(float v) {
    return 1.0f - 2.0f / (__expf(2.0f * v) + 1.0f);
}

// 16-byte global(bf16) -> LDS copy. Async (ASYNCcnt-tracked) when available.
__device__ __forceinline__ void copy16(const u16* __restrict__ g, u16* l) {
#if HAVE_ASYNC
    __builtin_amdgcn_global_load_async_to_lds_b128(
        (__attribute__((address_space(1))) v4i*)g,
        (__attribute__((address_space(3))) v4i*)l, 0, 0);
#else
    *reinterpret_cast<uint4*>(l) = *reinterpret_cast<const uint4*>(g);
#endif
}

template <int N>
__device__ __forceinline__ void wait_stage() {
#if HAVE_ASYNC
#if __has_builtin(__builtin_amdgcn_s_wait_asynccnt)
    __builtin_amdgcn_s_wait_asynccnt(N);
#else
    asm volatile("s_wait_asynccnt %0" :: "i"(N) : "memory");
#endif
#endif
}

// ---------------- Pass 1: elementwise f32 -> bf16 --------------------------
__global__ __launch_bounds__(256) void cvt_bf16_kernel(
    const float* __restrict__ src, u16* __restrict__ dst, int n4)
{
    const int i = blockIdx.x * 256 + threadIdx.x;
    if (i < n4) {
        const float4 v = reinterpret_cast<const float4*>(src)[i];
        uint2 p; p.x = pack2(v.x, v.y); p.y = pack2(v.z, v.w);
        reinterpret_cast<uint2*>(dst)[i] = p;
    }
}

// ---------------- Pass 2: fused WMMA GEMM + LSTM epilogue ------------------
// Stage one K-step tile (A: 128x32 bf16, B: 4 gates x 64x32 bf16) into LDS.
__device__ __forceinline__ void stage_tile(
    const u16* __restrict__ xb, const u16* __restrict__ hb,
    const u16* __restrict__ wb, u16* As, u16* Bs,
    int kstep, int blockM, int blockN, int tid)
{
    const int  k0 = kstep * TILE_K;
    const bool hp = (k0 >= 1024);
    const int  kk = k0 & 1023;
    const u16* A  = hp ? hb : xb;
    const u16* W  = wb + (hp ? (size_t)4 * 1024 * 1024 : 0);

    // A: 128 rows x 4 chunks of 8 bf16 (16B) = 512 chunk-copies, 2/thread
    #pragma unroll
    for (int i = 0; i < 2; ++i) {
        const int f   = tid + 256 * i;       // 0..511
        const int row = f >> 2;              // 0..127
        const int ch  = f & 3;               // 0..3
        copy16(A + (size_t)(blockM + row) * 1024 + kk + ch * 8,
               &As[row * LDS_STRIDE + ch * 8]);
    }
    // B: 4 gates x 64 rows x 4 chunks = 1024 chunk-copies, 4/thread
    #pragma unroll
    for (int g = 0; g < 4; ++g) {
        const int row = tid >> 2;            // 0..63
        const int ch  = tid & 3;             // 0..3
        copy16(W + (size_t)g * 1024 * 1024 + (size_t)(blockN + row) * 1024 + kk + ch * 8,
               &Bs[(g * TILE_N + row) * LDS_STRIDE + ch * 8]);
    }
}

__global__ __launch_bounds__(256) void lstm_wmma_kernel(
    const u16* __restrict__ xb, const u16* __restrict__ hb,
    const u16* __restrict__ wb,                       // [2 phases][4 gates][1024][1024]
    const float* __restrict__ cprev,
    const float* __restrict__ bxi, const float* __restrict__ bxf,
    const float* __restrict__ bxg, const float* __restrict__ bxo,
    const float* __restrict__ bhi, const float* __restrict__ bhf,
    const float* __restrict__ bhg, const float* __restrict__ bho,
    float* __restrict__ out)
{
    constexpr int BATCH  = 8192;
    constexpr int H      = 1024;
    constexpr int NSTEPS = 2048 / TILE_K;   // 64

    __shared__ u16 As[2 * A_BUF];           // 20 KB (double buffered)
    __shared__ u16 Bs[2 * B_BUF];           // 40 KB

    const int tid    = threadIdx.x;
    const int lane   = tid & 31;
    const int wave   = tid >> 5;
    const int waveM  = wave >> 2;           // 0..1  (64 rows each)
    const int waveN  = wave & 3;            // 0..3  (16 cols each)
    const int blockM = blockIdx.x * TILE_M;
    const int blockN = blockIdx.y * TILE_N;
    const int half   = (lane < 16) ? 0 : 1; // K-half owned by this lane group

    v8f acc[4][4];                          // [m_subtile][gate]
    #pragma unroll
    for (int m = 0; m < 4; ++m)
        #pragma unroll
        for (int g = 0; g < 4; ++g)
            acc[m][g] = (v8f){};

    // Prologue: stage K-step 0 into buffer 0
    stage_tile(xb, hb, wb, As, Bs, 0, blockM, blockN, tid);

    for (int ks = 0; ks < NSTEPS; ++ks) {
        const int cur = ks & 1;
        const int nxt = cur ^ 1;
        u16* Ac = As + cur * A_BUF;
        u16* Bc = Bs + cur * B_BUF;

        if (ks + 1 < NSTEPS) {
            stage_tile(xb, hb, wb, As + nxt * A_BUF, Bs + nxt * B_BUF,
                       ks + 1, blockM, blockN, tid);
            wait_stage<6>();                // 6 async ops just issued for nxt;
        } else {                            // in-order => cur's 6 have landed
            wait_stage<0>();
        }
        __syncthreads();

        // A 16x32 frag: lanes 0-15 hold K 0..7 & 16..23; lanes 16-31 the rest
        v16bf aF[4], bF[4];
        #pragma unroll
        for (int m = 0; m < 4; ++m) {
            const int r = waveM * 64 + m * 16 + (lane & 15);
            const u16* p = &Ac[r * LDS_STRIDE + half * 8];
            union { uint4 u[2]; v16bf v; } t;
            t.u[0] = *reinterpret_cast<const uint4*>(p);
            t.u[1] = *reinterpret_cast<const uint4*>(p + 16);
            aF[m] = t.v;
        }
        // B 32x16 frag: lanes 0-15 hold K 0..7 & 16..23 of col n; lanes 16-31 the rest
        #pragma unroll
        for (int g = 0; g < 4; ++g) {
            const int n = waveN * 16 + (lane & 15);
            const u16* p = &Bc[(g * TILE_N + n) * LDS_STRIDE + half * 8];
            union { uint4 u[2]; v16bf v; } t;
            t.u[0] = *reinterpret_cast<const uint4*>(p);
            t.u[1] = *reinterpret_cast<const uint4*>(p + 16);
            bF[g] = t.v;
        }

        #pragma unroll
        for (int m = 0; m < 4; ++m)
            #pragma unroll
            for (int g = 0; g < 4; ++g)
                acc[m][g] = __builtin_amdgcn_wmma_f32_16x16x32_bf16(
                    false, aF[m], false, bF[g], (short)0, acc[m][g], false, false);

        __syncthreads();   // protect cur buffer before it is restaged at ks+1
    }

    // ---- Epilogue: biases + activations + cell update ----
    const int jcol = blockN + waveN * 16 + (lane & 15);
    float bsum[4];
    bsum[0] = bxi[jcol] + bhi[jcol];
    bsum[1] = bxf[jcol] + bhf[jcol];
    bsum[2] = bxg[jcol] + bhg[jcol];
    bsum[3] = bxo[jcol] + bho[jcol];

    float* __restrict__ hOut = out;
    float* __restrict__ cOut = out + (size_t)BATCH * H;

    #pragma unroll
    for (int m = 0; m < 4; ++m) {
        #pragma unroll
        for (int r = 0; r < 8; ++r) {
            const int brow = blockM + waveM * 64 + m * 16 + r + ((lane >> 4) << 3);
            const size_t idx = (size_t)brow * H + jcol;
            const float gi = acc[m][0][r] + bsum[0];
            const float gf = acc[m][1][r] + bsum[1];
            const float gg = acc[m][2][r] + bsum[2];
            const float go = acc[m][3][r] + bsum[3];
            const float iv = fast_sigmoid(gi);
            const float fv = fast_sigmoid(gf);
            const float gv = fast_tanh(gg);
            const float ov = fast_sigmoid(go);
            const float cn = fv * cprev[idx] + iv * gv;
            cOut[idx] = cn;
            hOut[idx] = ov * fast_tanh(cn);
        }
    }
}

extern "C" void kernel_launch(void* const* d_in, const int* in_sizes, int n_in,
                              void* d_out, int out_size, void* d_ws, size_t ws_size,
                              hipStream_t stream) {
    (void)in_sizes; (void)n_in; (void)out_size; (void)ws_size;
    const float* x    = (const float*)d_in[0];
    const float* h    = (const float*)d_in[1];
    const float* c    = (const float*)d_in[2];
    const float* W_ii = (const float*)d_in[3];  const float* b_ii = (const float*)d_in[4];
    const float* W_hi = (const float*)d_in[5];  const float* b_hi = (const float*)d_in[6];
    const float* W_if = (const float*)d_in[7];  const float* b_if = (const float*)d_in[8];
    const float* W_hf = (const float*)d_in[9];  const float* b_hf = (const float*)d_in[10];
    const float* W_ig = (const float*)d_in[11]; const float* b_ig = (const float*)d_in[12];
    const float* W_hg = (const float*)d_in[13]; const float* b_hg = (const float*)d_in[14];
    const float* W_io = (const float*)d_in[15]; const float* b_io = (const float*)d_in[16];
    const float* W_ho = (const float*)d_in[17]; const float* b_ho = (const float*)d_in[18];

    // Workspace layout (bf16): xb[8M] | hb[8M] | W[2][4][1024*1024]  = 48 MB
    u16* xb = (u16*)d_ws;
    u16* hb = xb + (size_t)8 * 1024 * 1024;
    u16* wbase = hb + (size_t)8 * 1024 * 1024;
    const size_t WM = (size_t)1024 * 1024;

    auto cvt = [&](const float* s, u16* d, size_t n) {
        const int n4 = (int)(n / 4);
        cvt_bf16_kernel<<<(n4 + 255) / 256, 256, 0, stream>>>(s, d, n4);
    };
    cvt(x, xb, (size_t)8192 * 1024);
    cvt(h, hb, (size_t)8192 * 1024);
    cvt(W_ii, wbase + 0 * WM, WM);   // phase 0 (x-side): i, f, g, o
    cvt(W_if, wbase + 1 * WM, WM);
    cvt(W_ig, wbase + 2 * WM, WM);
    cvt(W_io, wbase + 3 * WM, WM);
    cvt(W_hi, wbase + 4 * WM, WM);   // phase 1 (h-side): i, f, g, o
    cvt(W_hf, wbase + 5 * WM, WM);
    cvt(W_hg, wbase + 6 * WM, WM);
    cvt(W_ho, wbase + 7 * WM, WM);

    dim3 grid(8192 / TILE_M, 1024 / TILE_N);   // 64 x 16 blocks
    lstm_wmma_kernel<<<grid, 256, 0, stream>>>(
        xb, hb, wbase, c,
        b_ii, b_if, b_ig, b_io,
        b_hi, b_hf, b_hg, b_ho,
        (float*)d_out);
}
